// POS_10479720202477
// MI455X (gfx1250) — compile-verified, matrix operand order checked by reading, not run on hardware
//
#include <hip/hip_runtime.h>

typedef __attribute__((ext_vector_type(16))) _Float16 v16h;
typedef __attribute__((ext_vector_type(8)))  float    v8f;

#define B_   8
#define C_   3
#define N_   900
#define HW_  16384   // 128*128
#define L_   48
#define K_   852     // N - L

// ---------------------------------------------------------------------------
// Stage 1: spatial mean of each (b,c,n) 128x128 slice via f16 ones-GEMM WMMA.
// One 256-thread block (8 wave32) per slice. Each wave consumes 2048 floats:
// 4 chained v_wmma_f32_16x16x32_f16, A = data (f16), B = ones, C = f32 accum.
// Sum of all D elements = 16 * sum of fed elements (16 replicated columns).
// ---------------------------------------------------------------------------
__global__ __launch_bounds__(256) void spatial_mean_wmma(
    const float* __restrict__ x, float* __restrict__ rgb) {
  const int slice = blockIdx.x;            // ((b*3 + c)*900 + n)
  const int tid   = threadIdx.x;
  const int wave  = tid >> 5;
  const int lane  = tid & 31;

  const float4* px = (const float4*)x + (size_t)slice * (HW_ / 4) + wave * (2048 / 4);

  // Prefetch second half of this wave's region (gfx1250 global_prefetch_b8).
  __builtin_prefetch(px + 256 + lane, 0, 3);

  v16h ones;
#pragma unroll
  for (int i = 0; i < 16; ++i) ones[i] = (_Float16)1.0f;

  v8f c = {0.f, 0.f, 0.f, 0.f, 0.f, 0.f, 0.f, 0.f};

#pragma unroll
  for (int it = 0; it < 4; ++it) {
    v16h a;
#pragma unroll
    for (int q = 0; q < 4; ++q) {
      float4 v = px[it * 128 + q * 32 + lane];   // coalesced 512B / wave / load
      a[q * 4 + 0] = (_Float16)v.x;
      a[q * 4 + 1] = (_Float16)v.y;
      a[q * 4 + 2] = (_Float16)v.z;
      a[q * 4 + 3] = (_Float16)v.w;
    }
    c = __builtin_amdgcn_wmma_f32_16x16x32_f16(
        /*neg_a=*/false, a, /*neg_b=*/false, ones,
        /*c_mod=*/(short)0, c, /*reuse_a=*/false, /*reuse_b=*/false);
  }

  float s = c[0] + c[1] + c[2] + c[3] + c[4] + c[5] + c[6] + c[7];

  __shared__ float red[256];
  red[tid] = s;
  __syncthreads();
#pragma unroll
  for (int off = 128; off > 0; off >>= 1) {
    if (tid < off) red[tid] += red[tid + off];
    __syncthreads();
  }
  if (tid == 0) {
    float mean = red[0] * (1.0f / (16.0f * (float)HW_));
    int b  = slice / (C_ * N_);
    int r  = slice % (C_ * N_);
    int ch = r / N_;
    int n  = r % N_;
    rgb[((size_t)b * N_ + n) * 3 + ch] = mean;   // (B, N, 3) layout
  }
}

// ---------------------------------------------------------------------------
// Stage 2: POS per sliding window. One thread per (b,k). L2-resident data.
//   Cn = W / mean_t(W);  S0 = G-B;  S1 = G+B-2R;  alpha = std(S0)/std(S1);
//   h = S0 + alpha*S1 - mean(h)
// ---------------------------------------------------------------------------
__global__ __launch_bounds__(256) void pos_windows(
    const float* __restrict__ rgb, float* __restrict__ hbuf) {
  int idx = blockIdx.x * 256 + threadIdx.x;
  if (idx >= B_ * K_) return;
  int b = idx / K_;
  int k = idx % K_;
  const float* r = rgb + ((size_t)b * N_ + k) * 3;

  float sr = 0.f, sg = 0.f, sb = 0.f;
#pragma unroll 4
  for (int j = 0; j < L_; ++j) {
    sr += r[j * 3 + 0]; sg += r[j * 3 + 1]; sb += r[j * 3 + 2];
  }
  float ir = (float)L_ / sr, ig = (float)L_ / sg, ib = (float)L_ / sb;

  float s0 = 0.f, q0 = 0.f, s1 = 0.f, q1 = 0.f;
#pragma unroll 4
  for (int j = 0; j < L_; ++j) {
    float cr = r[j * 3 + 0] * ir, cg = r[j * 3 + 1] * ig, cb = r[j * 3 + 2] * ib;
    float S0 = cg - cb;
    float S1 = cg + cb - 2.0f * cr;
    s0 += S0; q0 += S0 * S0;
    s1 += S1; q1 += S1 * S1;
  }
  float m0 = s0 * (1.0f / L_), m1 = s1 * (1.0f / L_);
  float v0 = fmaxf(q0 - (float)L_ * m0 * m0, 0.0f) * (1.0f / (L_ - 1));
  float v1 = fmaxf(q1 - (float)L_ * m1 * m1, 0.0f) * (1.0f / (L_ - 1));
  float alpha = sqrtf(v0 / v1);
  float mh = m0 + alpha * m1;

  float* hout = hbuf + (size_t)idx * L_;
#pragma unroll 4
  for (int j = 0; j < L_; ++j) {
    float cr = r[j * 3 + 0] * ir, cg = r[j * 3 + 1] * ig, cb = r[j * 3 + 2] * ib;
    float S0 = cg - cb;
    float S1 = cg + cb - 2.0f * cr;
    hout[j] = S0 + alpha * S1 - mh;
  }
}

// ---------------------------------------------------------------------------
// Stage 3: deterministic overlap-add as a gather: H[b,n] = sum_k h[b,k,n-k].
// ---------------------------------------------------------------------------
__global__ __launch_bounds__(256) void overlap_add(
    const float* __restrict__ hbuf, float* __restrict__ H) {
  int idx = blockIdx.x * 256 + threadIdx.x;
  if (idx >= B_ * N_) return;
  int b = idx / N_;
  int n = idx % N_;
  int k0 = n - (L_ - 1); if (k0 < 0) k0 = 0;
  int k1 = n;            if (k1 > K_ - 1) k1 = K_ - 1;
  float acc = 0.0f;
  for (int k = k0; k <= k1; ++k)
    acc += hbuf[((size_t)b * K_ + k) * L_ + (n - k)];
  H[idx] = acc;
}

extern "C" void kernel_launch(void* const* d_in, const int* in_sizes, int n_in,
                              void* d_out, int out_size, void* d_ws, size_t ws_size,
                              hipStream_t stream) {
  const float* x = (const float*)d_in[0];   // (8,3,900,128,128) f32
  float* H = (float*)d_out;                 // (8,900) f32

  // Workspace: rgb (B,N,3) = 21600 f32 padded to 21632 for 256B alignment,
  // then h (B,K,L) = 327168 f32.  Total ~1.4 MB.
  float* rgb  = (float*)d_ws;
  float* hbuf = (float*)d_ws + 21632;

  spatial_mean_wmma<<<B_ * C_ * N_, 256, 0, stream>>>(x, rgb);
  pos_windows<<<(B_ * K_ + 255) / 256, 256, 0, stream>>>(rgb, hbuf);
  overlap_add<<<(B_ * N_ + 255) / 256, 256, 0, stream>>>(hbuf, H);
}
